// BidafAttention_39977555591439
// MI455X (gfx1250) — matrix operand order repaired and checked by reading, not compiled, not run on hardware
//
#include <hip/hip_runtime.h>
#include <hip/hip_bf16.h>

typedef __attribute__((ext_vector_type(16))) _Float16 v16h;
typedef __attribute__((ext_vector_type(8)))  float    v8f;
typedef __attribute__((ext_vector_type(4)))  unsigned int u32x4;
typedef __attribute__((ext_vector_type(8)))  int      i32x8;

#define NB   16
#define LL   1024
#define DD   1024
#define LDSTR 40   // 32 halves data + 8 halves (16B) pad per row; 80B rows keep 16B align

union FragV {
    u32x4    u[2];
    v16h     v;
    _Float16 h[16];
};

// ---------------------------------------------------------------------------
// TDM: DMA one [64 rows x 32 halves] tile (row stride `ld` halves) from global
// into LDS, writing rows at LDSTR(=40 half) pitch via D# padding:
//   pad_enable=1, pad_interval=3 (after 16 DWORDs = 64B), pad_amount=3 (4 DW=16B)
// 2D tensor -> 2-operand form (VADDR2/3 = NULL).  Tracked by TENSORcnt.
// ---------------------------------------------------------------------------
static __device__ __forceinline__ void tdm_load_tile(
    const _Float16* gtile, const void* lds, unsigned ld)
{
    unsigned long long ga = (unsigned long long)(size_t)gtile;
    unsigned ldsb = (unsigned)(size_t)lds;      // low 32 bits of flat = LDS byte addr

    u32x4 g0;
    g0[0] = 1u;                                  // count=1, user mode, no gather
    g0[1] = ldsb;                                // lds_addr [63:32]
    g0[2] = (unsigned)ga;                        // global_addr [95:64]
    g0[3] = (unsigned)(ga >> 32) | (2u << 30);   // global_addr[56:32] | type=2

    unsigned td0 = ld;                           // tensor_dim0 (elements)
    unsigned td1 = 1u << 20;                     // tensor_dim1 (huge, no OOB)
    i32x8 g1;
    g1[0] = (int)((1u << 16) |                   // data_size = 2 bytes
                  (1u << 20) |                   // pad_enable
                  (3u << 22) |                   // pad_interval: 16 DWORDs
                  (3u << 25));                   // pad_amount : 4 DWORDs
    g1[1] = (int)((td0 & 0xFFFFu) << 16);        // [63:48] = tensor_dim0 lo16
    g1[2] = (int)((td0 >> 16) | ((td1 & 0xFFFFu) << 16));
    g1[3] = (int)((td1 >> 16) | (32u << 16));    // tile_dim0 = 32  [127:112]
    g1[4] = (int)(64u);                          // tile_dim1 = 64, tile_dim2 = 0
    g1[5] = (int)ld;                             // tensor_dim0_stride lo32
    g1[6] = 0;                                   // stride hi16 | dim1_stride lo16
    g1[7] = 0;

    asm volatile("tensor_load_to_lds %0, %1" :: "s"(g0), "s"(g1) : "memory");
}

// ---------------------------------------------------------------------------
// Pass 0: ul = lhs@w_l, ur = rhs@w_r; f16 copies of lhs*w_prod, lhs, rhs.
// ---------------------------------------------------------------------------
__global__ __launch_bounds__(256) void prep_kernel(
    const float* __restrict__ lhs, const float* __restrict__ rhs,
    const float* __restrict__ w,
    _Float16* __restrict__ lhsw_h, _Float16* __restrict__ lhs_h,
    _Float16* __restrict__ rhs_h,
    float* __restrict__ ul, float* __restrict__ ur)
{
    __shared__ float red[256];
    int row = blockIdx.x;
    bool isL = row < NB * LL;
    int r = isL ? row : row - NB * LL;
    const float* src  = (isL ? lhs : rhs) + (long)r * DD;
    const float* wvec = isL ? (w + DD) : (w + 2 * DD);
    int t = threadIdx.x;

    float4 x  = *(const float4*)(src  + t * 4);
    float4 wv = *(const float4*)(wvec + t * 4);
    float partial = x.x * wv.x + x.y * wv.y + x.z * wv.z + x.w * wv.w;

    if (isL) {
        float4 wp = *(const float4*)(w + t * 4);
        _Float16* d1 = lhsw_h + (long)r * DD + t * 4;
        d1[0] = (_Float16)(x.x * wp.x); d1[1] = (_Float16)(x.y * wp.y);
        d1[2] = (_Float16)(x.z * wp.z); d1[3] = (_Float16)(x.w * wp.w);
        _Float16* d2 = lhs_h + (long)r * DD + t * 4;
        d2[0] = (_Float16)x.x; d2[1] = (_Float16)x.y;
        d2[2] = (_Float16)x.z; d2[3] = (_Float16)x.w;
    } else {
        _Float16* d = rhs_h + (long)r * DD + t * 4;
        d[0] = (_Float16)x.x; d[1] = (_Float16)x.y;
        d[2] = (_Float16)x.z; d[3] = (_Float16)x.w;
    }

    red[t] = partial;
    __syncthreads();
    for (int s = 128; s > 0; s >>= 1) {
        if (t < s) red[t] += red[t + s];
        __syncthreads();
    }
    if (t == 0) (isL ? ul : ur)[r] = red[0];
}

// ---------------------------------------------------------------------------
// Tiled f16 transpose (per batch 1024x1024): out[d][r] = in[r][d].
// ---------------------------------------------------------------------------
__global__ __launch_bounds__(256) void transpose_f16_kernel(
    const _Float16* __restrict__ in, _Float16* __restrict__ outp)
{
    __shared__ __align__(16) _Float16 T[64 * 72];
    int b   = blockIdx.z;
    int gr0 = blockIdx.x * 64;   // source row block
    int gc0 = blockIdx.y * 64;   // source col block
    const _Float16* src = in   + (long)b * LL * DD;
    _Float16*       dst = outp + (long)b * LL * DD;
    int tid = threadIdx.x;

    #pragma unroll
    for (int it = 0; it < 2; ++it) {
        int c = tid + it * 256;
        int r = c >> 3, col = (c & 7) * 8;
        *(u32x4*)&T[r * 72 + col] =
            *(const u32x4*)(src + (long)(gr0 + r) * DD + gc0 + col);
    }
    __syncthreads();
    #pragma unroll
    for (int it = 0; it < 2; ++it) {
        int c = tid + it * 256;
        int dcol = c >> 3;          // output row (d)
        int rcol = (c & 7) * 8;     // 8 consecutive source rows
        union { u32x4 u; _Float16 h[8]; } v;
        #pragma unroll
        for (int i = 0; i < 8; ++i) v.h[i] = T[(rcol + i) * 72 + dcol];
        *(u32x4*)(dst + (long)(gc0 + dcol) * LL + gr0 + rcol) = v.u;
    }
}

// ---------------------------------------------------------------------------
// TDM double-buffered WMMA GEMM: C(MxN) = A(MxK) * B, B given as B^T rows
// ([N][K]).  64x64 tile, 4 waves, BK=32.  Wave 0 drives the Tensor Data
// Mover; s_wait_tensorcnt(2) overlaps DMA of tile i+1 with WMMA on tile i.
// EPI=0: C = tanh(acc + ul[m] + ur[n] + bias);  EPI=1: C = acc.
// ---------------------------------------------------------------------------
template<int EPI>
__global__ __launch_bounds__(128) void gemm_wmma_kernel(
    const _Float16* __restrict__ A, int lda, long strideA,
    const _Float16* __restrict__ Bt, int ldb, long strideB,
    float* __restrict__ C, int ldc, long strideC,
    int K,
    const float* __restrict__ ul, const float* __restrict__ ur,
    const float* __restrict__ bptr)
{
    __shared__ __align__(16) _Float16 As[2][64 * LDSTR];
    __shared__ __align__(16) _Float16 Bs[2][64 * LDSTR];
    __shared__ float ulS[64], urS[64];

    const int b   = blockIdx.z;
    const int gm0 = blockIdx.x * 64;
    const int gn0 = blockIdx.y * 64;
    const _Float16* Ab = A  + (long)b * strideA + (long)gm0 * lda;
    const _Float16* Bb = Bt + (long)b * strideB + (long)gn0 * ldb;
    C += (long)b * strideC;

    const int tid  = threadIdx.x;
    const int lane = tid & 31;
    const int wave = tid >> 5;

    float bias = 0.0f;
    if constexpr (EPI == 0) {
        bias = bptr[0];
        if (tid < 64)       ulS[tid]      = ul[(long)b * LL + gm0 + tid];
        else                urS[tid - 64] = ur[(long)b * LL + gn0 + (tid - 64)];
    }

    v8f acc[4] = {};
    const int m16 = lane & 15;
    const int kh  = lane >> 4;
    const int TILES = K / 32;

    if (wave == 0) {                       // prologue: DMA tile 0
        tdm_load_tile(Ab, &As[0][0], (unsigned)lda);
        tdm_load_tile(Bb, &Bs[0][0], (unsigned)ldb);
    }

    for (int i = 0; i < TILES; ++i) {
        const int cur = i & 1;
        if (wave == 0) {
            if (i + 1 < TILES) {           // DMA tile i+1 into other buffer
                tdm_load_tile(Ab + (i + 1) * 32, &As[cur ^ 1][0], (unsigned)lda);
                tdm_load_tile(Bb + (i + 1) * 32, &Bs[cur ^ 1][0], (unsigned)ldb);
                __builtin_amdgcn_s_wait_tensorcnt(2);   // tile i landed
            } else {
                __builtin_amdgcn_s_wait_tensorcnt(0);
            }
        }
        __syncthreads();                   // publish LDS tile i to all waves

        // A fragment (ISA 16-bit 16x32 layout)
        FragV fa;
        const _Float16* ar = &As[cur][(wave * 16 + m16) * LDSTR];
        fa.u[0] = *(const u32x4*)(ar + kh * 8);
        fa.u[1] = *(const u32x4*)(ar + 16 + kh * 8);

        #pragma unroll
        for (int j = 0; j < 4; ++j) {
            FragV fb;   // B fragment: same K order as A (ISA 16-bit layout)
            const _Float16* br = &Bs[cur][(j * 16 + m16) * LDSTR];
            fb.u[0] = *(const u32x4*)(br + kh * 8);
            fb.u[1] = *(const u32x4*)(br + 16 + kh * 8);
            acc[j] = __builtin_amdgcn_wmma_f32_16x16x32_f16(
                false, fa.v, false, fb.v, (short)0, acc[j], false, false);
        }
        __syncthreads();                   // done reading before TDM overwrites
    }

    // epilogue: C/D layout lanes0-15 M=v, lanes16-31 M=8+v
    const int mrow  = wave * 16 + (lane >> 4) * 8;
    const int ncol0 = lane & 15;
    #pragma unroll
    for (int j = 0; j < 4; ++j) {
        int gn = gn0 + j * 16 + ncol0;
        #pragma unroll
        for (int v = 0; v < 8; ++v) {
            int gm = gm0 + mrow + v;
            float x = acc[j][v];
            if constexpr (EPI == 0)
                x = tanhf(x + ulS[mrow + v] + urS[j * 16 + ncol0] + bias);
            C[(long)gm * ldc + gn] = x;
        }
    }
}

// ---------------------------------------------------------------------------
// Softmax over axis 2 (r): one block per (b,l) row -> normalized f16 a_lhs.
// ---------------------------------------------------------------------------
__global__ __launch_bounds__(256) void softmax_rows_kernel(
    const float* __restrict__ S, _Float16* __restrict__ E)
{
    __shared__ float red[256];
    long row = blockIdx.x;
    const float* s = S + row * LL;
    int t = threadIdx.x;

    float4 v = *(const float4*)(s + t * 4);
    red[t] = fmaxf(fmaxf(v.x, v.y), fmaxf(v.z, v.w));
    __syncthreads();
    for (int k = 128; k > 0; k >>= 1) {
        if (t < k) red[t] = fmaxf(red[t], red[t + k]);
        __syncthreads();
    }
    float mx = red[0];
    __syncthreads();

    float e0 = __expf(v.x - mx), e1 = __expf(v.y - mx);
    float e2 = __expf(v.z - mx), e3 = __expf(v.w - mx);
    red[t] = e0 + e1 + e2 + e3;
    __syncthreads();
    for (int k = 128; k > 0; k >>= 1) {
        if (t < k) red[t] += red[t + k];
        __syncthreads();
    }
    float inv = 1.0f / red[0];

    _Float16* d = E + row * LL + t * 4;
    d[0] = (_Float16)(e0 * inv); d[1] = (_Float16)(e1 * inv);
    d[2] = (_Float16)(e2 * inv); d[3] = (_Float16)(e3 * inv);
}

// ---------------------------------------------------------------------------
// Softmax over axis 1 (l): one block per (b,r) column -> normalized f16
// a_rhs^T (written as rows of l, coalesced).
// ---------------------------------------------------------------------------
__global__ __launch_bounds__(256) void softmax_cols_kernel(
    const float* __restrict__ S, _Float16* __restrict__ ET)
{
    __shared__ float red[256];
    int col = blockIdx.x;
    int b = col >> 10, r = col & 1023;
    const float* s = S + (long)b * LL * LL + r;
    int t = threadIdx.x;

    float v0 = s[(long)(t)       * LL];
    float v1 = s[(long)(t + 256) * LL];
    float v2 = s[(long)(t + 512) * LL];
    float v3 = s[(long)(t + 768) * LL];

    red[t] = fmaxf(fmaxf(v0, v1), fmaxf(v2, v3));
    __syncthreads();
    for (int k = 128; k > 0; k >>= 1) {
        if (t < k) red[t] = fmaxf(red[t], red[t + k]);
        __syncthreads();
    }
    float mx = red[0];
    __syncthreads();

    float e0 = __expf(v0 - mx), e1 = __expf(v1 - mx);
    float e2 = __expf(v2 - mx), e3 = __expf(v3 - mx);
    red[t] = e0 + e1 + e2 + e3;
    __syncthreads();
    for (int k = 128; k > 0; k >>= 1) {
        if (t < k) red[t] += red[t + k];
        __syncthreads();
    }
    float inv = 1.0f / red[0];

    _Float16* d = ET + (long)col * LL;
    d[t]       = (_Float16)(e0 * inv);
    d[t + 256] = (_Float16)(e1 * inv);
    d[t + 512] = (_Float16)(e2 * inv);
    d[t + 768] = (_Float16)(e3 * inv);
}

// ---------------------------------------------------------------------------
// Identity halves: lhs -> lhs_out[:,:,0:D], rhs -> rhs_out[:,:,0:D].
// ---------------------------------------------------------------------------
__global__ __launch_bounds__(256) void copy_halves_kernel(
    const float* __restrict__ lhs, const float* __restrict__ rhs,
    float* __restrict__ out)
{
    int row = blockIdx.x;
    bool isL = row < NB * LL;
    int r = isL ? row : row - NB * LL;
    const float* src = (isL ? lhs : rhs) + (long)r * DD;
    float* dst = out + (isL ? 0L : (long)NB * LL * 2 * DD) + (long)r * (2 * DD);
    int t = threadIdx.x;
    *(float4*)(dst + t * 4) = *(const float4*)(src + t * 4);
}

// ---------------------------------------------------------------------------
extern "C" void kernel_launch(void* const* d_in, const int* in_sizes, int n_in,
                              void* d_out, int out_size, void* d_ws, size_t ws_size,
                              hipStream_t stream)
{
    const float* lhs = (const float*)d_in[0];   // [16,1024,1024]
    const float* rhs = (const float*)d_in[1];   // [16,1024,1024]
    const float* w   = (const float*)d_in[2];   // [3072]
    const float* bpt = (const float*)d_in[3];   // scalar
    float* out = (float*)d_out;                 // [2][16][1024][2048]

    char* ws = (char*)d_ws;
    const long EL16 = (long)NB * LL * DD;                    // 16M elems
    _Float16* lhsw_h = (_Float16*)(ws);                      //  32 MB
    _Float16* lhs_h  = (_Float16*)(ws + EL16 * 2);           //  32 MB
    _Float16* rhs_h  = (_Float16*)(ws + EL16 * 4);           //  32 MB
    float*    S      = (float*)   (ws + EL16 * 6);           //  64 MB
    _Float16* Elhs   = (_Float16*)(ws + EL16 * 10);          //  32 MB
    _Float16* ErT    = (_Float16*)(ws + EL16 * 12);          //  32 MB
    _Float16* rhs_hT = (_Float16*)(ws + EL16 * 14);          //  32 MB
    _Float16* lhs_hT = (_Float16*)(ws + EL16 * 16);          //  32 MB
    float*    ul     = (float*)   (ws + EL16 * 18);          //  64 KB
    float*    ur     = (float*)   (ws + EL16 * 18 + (long)NB * LL * 4);

    const long sAB = (long)LL * DD;        // per-batch stride, f16 mats
    const long sS  = (long)LL * LL;        // per-batch stride, score (f32)
    const long sO  = (long)LL * 2 * DD;    // per-batch stride, output (f32)

    // 0) row dots + f16 conversions
    prep_kernel<<<2 * NB * LL, 256, 0, stream>>>(
        lhs, rhs, w, lhsw_h, lhs_h, rhs_h, ul, ur);

    // 0b) f16 transposes so every GEMM consumes [N][K] B operands
    transpose_f16_kernel<<<dim3(16, 16, NB), 256, 0, stream>>>(rhs_h, rhs_hT);
    transpose_f16_kernel<<<dim3(16, 16, NB), 256, 0, stream>>>(lhs_h, lhs_hT);

    // 1) score = tanh((lhs*w_prod) @ rhs^T + ul + ur + b);  B^T rows = rhs rows
    gemm_wmma_kernel<0><<<dim3(16, 16, NB), 128, 0, stream>>>(
        lhsw_h, DD, sAB, rhs_h, DD, sAB, S, LL, sS, DD, ul, ur, bpt);

    // 2) softmax over r -> normalized a_lhs (f16)
    softmax_rows_kernel<<<NB * LL, 256, 0, stream>>>(S, Elhs);

    // 3) softmax over l -> normalized a_rhs^T (f16, transposed)
    softmax_cols_kernel<<<NB * LL, 256, 0, stream>>>(S, ErT);

    // 4) lhs_out[:,:,D:2D] = a_lhs @ rhs;   B^T rows = rhs^T rows = rhs_hT
    gemm_wmma_kernel<1><<<dim3(16, 16, NB), 128, 0, stream>>>(
        Elhs, LL, sAB, rhs_hT, LL, sAB, out + DD, 2 * DD, sO, LL,
        nullptr, nullptr, nullptr);

    // 5) rhs_out[:,:,D:2D] = a_rhs^T @ lhs; B^T rows = lhs^T rows = lhs_hT
    gemm_wmma_kernel<1><<<dim3(16, 16, NB), 128, 0, stream>>>(
        ErT, LL, sAB, lhs_hT, LL, sAB,
        out + (long)NB * LL * 2 * DD + DD, 2 * DD, sO, LL,
        nullptr, nullptr, nullptr);

    // 6) identity halves
    copy_halves_kernel<<<2 * NB * LL, 256, 0, stream>>>(lhs, rhs, out);
}
